// SelfAttention_67061619360018
// MI455X (gfx1250) — compile-verified, matrix operand order checked by reading, not run on hardware
//
#include <hip/hip_runtime.h>
#include <hip/hip_bf16.h>
#include <math.h>

// ---------------------------------------------------------------------------
// Shapes (fixed by the reference): B=8, T=2048, H=D=1024, band width 512
// ---------------------------------------------------------------------------
#define BATCH 8
#define SEQ   2048
#define DIM   1024
#define BAND_LO 256   // keep if q - k <= 256
#define BAND_HI 255   // keep if k - q <= 255
#define INV_SQRT_D 0.03125f   // 1/sqrt(1024)

typedef __attribute__((ext_vector_type(16))) __bf16 v16bf;
typedef __attribute__((ext_vector_type(8)))  __bf16 v8bf;
typedef __attribute__((ext_vector_type(2)))  __bf16 v2bf;
typedef __attribute__((ext_vector_type(8)))  float  v8f;
typedef __attribute__((ext_vector_type(4)))  float  v4f;
typedef __attribute__((ext_vector_type(4)))  uint32_t v4u;
typedef __attribute__((ext_vector_type(8)))  uint32_t v8u;

__device__ inline v8f zero8() {
    v8f z;
#pragma unroll
    for (int i = 0; i < 8; ++i) z[i] = 0.0f;
    return z;
}

__device__ inline v8f wmma_bf16(v16bf a, v16bf b, v8f c) {
    // D = A(16x32 bf16) * B(32x16 bf16) + C(16x16 f32)
    return __builtin_amdgcn_wmma_f32_16x16x32_bf16(
        /*neg_a=*/false, a, /*neg_b=*/false, b,
        /*c_mod=*/(short)0, c, /*reuse_a=*/false, /*reuse_b=*/false);
}

// Load one 16x32 bf16 fragment (A) or 32x16 fragment (B, transposed view).
// Per CDNA5 ISA 16-bit matrix layout: lane&15 selects the M/N row, each lane
// holds two contiguous 8-element runs along the contraction dim at
// c0 + (lane>>4)*8 and +16.  base is row-major [row][contraction], ld elems.
__device__ inline v16bf load_frag(const __bf16* base, int ld, int row, int c0, int lane) {
    const __bf16* p = base + (size_t)row * ld + c0 + ((lane >> 4) << 3);
    v8bf lo = *reinterpret_cast<const v8bf*>(p);
    v8bf hi = *reinterpret_cast<const v8bf*>(p + 16);
    v16bf f;
#pragma unroll
    for (int i = 0; i < 8; ++i) { f[i] = lo[i]; f[i + 8] = hi[i]; }
    return f;
}

// ---------------------------------------------------------------------------
// Kernel 1: QKV projection GEMM.  X[f32, (B*T)x1024] * W[f32, 1024x1024]
//   z==0 -> Q  bf16 [B*T][1024]
//   z==1 -> K  bf16 [B*T][1024]
//   z==2 -> Vt bf16 [B][1024][2048]   (transposed for the PV matmul)
// Tile: 128M x 128N x 32K, 8 waves (4x2), wave tile 32x64 (2x4 wmma tiles).
// ---------------------------------------------------------------------------
__global__ __launch_bounds__(256)
void qkv_gemm_kernel(const float* __restrict__ X,
                     const float* __restrict__ WQ,
                     const float* __restrict__ WK,
                     const float* __restrict__ WV,
                     __bf16* __restrict__ Qg,
                     __bf16* __restrict__ Kg,
                     __bf16* __restrict__ Vtg) {
    __shared__ __align__(16) __bf16 Xs[128 * 32];   // [m][k]
    __shared__ __align__(16) __bf16 Ws[128 * 32];   // [n][k] (transposed stage)

    const int z   = blockIdx.z;
    const float* W = (z == 0) ? WQ : (z == 1) ? WK : WV;
    __bf16* outp   = (z == 0) ? Qg : (z == 1) ? Kg : Vtg;
    const bool transposeOut = (z == 2);

    const int tid  = threadIdx.x;
    const int lane = tid & 31;
    const int wid  = tid >> 5;
    const int wm   = wid >> 1;          // 0..3 -> 32-row slabs
    const int wn   = wid & 1;           // 0..1 -> 64-col slabs
    const int gn0  = blockIdx.x * 128;
    const int gm0  = blockIdx.y * 128;

    v8f acc[2][4];
#pragma unroll
    for (int mt = 0; mt < 2; ++mt)
#pragma unroll
        for (int nt = 0; nt < 4; ++nt) acc[mt][nt] = zero8();

    // Staging index precompute (vectorized float4 loads).
    const int xrow = tid >> 1;            // 0..127
    const int xc0  = (tid & 1) * 16;      // 0 or 16
    const int wk   = tid >> 3;            // 0..31
    const int wc0  = (tid & 7) * 16;      // 0..112

    for (int k0 = 0; k0 < DIM; k0 += 32) {
        // ---- Stage X tile 128x32 (fp32 -> bf16), b128 loads + b128 stores ----
        {
            const float* xp = &X[(size_t)(gm0 + xrow) * DIM + k0 + xc0];
            v4f x0 = *reinterpret_cast<const v4f*>(xp);
            v4f x1 = *reinterpret_cast<const v4f*>(xp + 4);
            v4f x2 = *reinterpret_cast<const v4f*>(xp + 8);
            v4f x3 = *reinterpret_cast<const v4f*>(xp + 12);
            v8bf b0, b1;
#pragma unroll
            for (int i = 0; i < 4; ++i) {
                b0[i] = (__bf16)x0[i]; b0[i + 4] = (__bf16)x1[i];
                b1[i] = (__bf16)x2[i]; b1[i + 4] = (__bf16)x3[i];
            }
            *reinterpret_cast<v8bf*>(&Xs[xrow * 32 + xc0])     = b0;
            *reinterpret_cast<v8bf*>(&Xs[xrow * 32 + xc0 + 8]) = b1;
        }
        // ---- Stage W tile 32x128 transposed into Ws[n][k] ----
        {
            const float* wp = &W[(size_t)(k0 + wk) * DIM + gn0 + wc0];
            v4f w0 = *reinterpret_cast<const v4f*>(wp);
            v4f w1 = *reinterpret_cast<const v4f*>(wp + 4);
            v4f w2 = *reinterpret_cast<const v4f*>(wp + 8);
            v4f w3 = *reinterpret_cast<const v4f*>(wp + 12);
#pragma unroll
            for (int i = 0; i < 4; ++i) {
                Ws[(wc0 + i)      * 32 + wk] = (__bf16)w0[i];
                Ws[(wc0 + 4 + i)  * 32 + wk] = (__bf16)w1[i];
                Ws[(wc0 + 8 + i)  * 32 + wk] = (__bf16)w2[i];
                Ws[(wc0 + 12 + i) * 32 + wk] = (__bf16)w3[i];
            }
        }
        __syncthreads();

#pragma unroll
        for (int mt = 0; mt < 2; ++mt) {
            const int mrow = wm * 32 + mt * 16 + (lane & 15);
            v16bf af = load_frag(Xs, 32, mrow, 0, lane);
#pragma unroll
            for (int nt = 0; nt < 4; ++nt) {
                const int nrow = wn * 64 + nt * 16 + (lane & 15);
                v16bf bf = load_frag(Ws, 32, nrow, 0, lane);
                acc[mt][nt] = wmma_bf16(af, bf, acc[mt][nt]);
            }
        }
        __syncthreads();
    }

    // ---- Store C (f32 accum -> bf16).  M = r + 8*(lane>>4), N = lane&15. ----
    if (!transposeOut) {
#pragma unroll
        for (int mt = 0; mt < 2; ++mt)
#pragma unroll
            for (int nt = 0; nt < 4; ++nt) {
                const int mbase = gm0 + wm * 32 + mt * 16 + ((lane >> 4) << 3);
                const int ncol  = gn0 + wn * 64 + nt * 16 + (lane & 15);
#pragma unroll
                for (int r = 0; r < 8; ++r)
                    outp[(size_t)(mbase + r) * DIM + ncol] = (__bf16)acc[mt][nt][r];
            }
    } else {
        // Consecutive r == consecutive t in Vt[b][d][t]: pack pairs -> b32.
#pragma unroll
        for (int mt = 0; mt < 2; ++mt)
#pragma unroll
            for (int nt = 0; nt < 4; ++nt) {
                const int mbase = gm0 + wm * 32 + mt * 16 + ((lane >> 4) << 3);
                const int ncol  = gn0 + wn * 64 + nt * 16 + (lane & 15);
                const int bb = mbase >> 11;          // / SEQ (8-run never crosses)
                const int t0 = mbase & (SEQ - 1);
                __bf16* vrow = &outp[((size_t)bb * DIM + ncol) * SEQ + t0];
#pragma unroll
                for (int r = 0; r < 8; r += 2) {
                    v2bf pr;
                    pr[0] = (__bf16)acc[mt][nt][r];
                    pr[1] = (__bf16)acc[mt][nt][r + 1];
                    *reinterpret_cast<v2bf*>(vrow + r) = pr;
                }
            }
    }
}

// ---------------------------------------------------------------------------
// Kernel 2: fused masked-softmax attention (flash style, dense over T).
// One workgroup = 16 queries of one batch; 8 waves split D into 128-wide
// slices.  The 16x1024 bf16 Q tile is DMA'd into LDS once by the Tensor
// Data Mover (tensor_load_to_lds) and reused for all 64 key steps.
// Per 32-key step: waves compute partial S over their d-slice (WMMA),
// reduce via ds_add_f32, 16 row-owner threads run the online softmax
// (masked entries contribute s=0, exactly like the reference), then every
// wave rescales its O slice and accumulates P*V via WMMA.
// ---------------------------------------------------------------------------
__global__ __launch_bounds__(256)
void attn_kernel(const __bf16* __restrict__ Qg,
                 const __bf16* __restrict__ Kg,
                 const __bf16* __restrict__ Vtg,
                 float* __restrict__ Og) {
    __shared__ __align__(16) __bf16 Qs[16 * DIM];       // 32 KB, TDM target
    __shared__ __align__(16) float  S_sh[16 * 32];
    __shared__ __align__(16) __bf16 P_sh[16 * 32];
    __shared__ float m_sh[16], l_sh[16], alpha_sh[16];

    const int tid   = threadIdx.x;
    const int lane  = tid & 31;
    const int wid   = tid >> 5;
    const int dbase = wid * 128;                    // this wave's D slice
    const int q0    = (blockIdx.x & 127) * 16;      // query tile in sequence
    const int b     = blockIdx.x >> 7;              // batch
    const size_t rowQ = (size_t)b * SEQ + q0;       // flattened Q row base

    // ---- TDM: DMA Q tile [16 x 1024] bf16 from global into LDS (wave 0) ----
    if (wid == 0) {
        const uint64_t gaddr = (uint64_t)(uintptr_t)(Qg + rowQ * DIM);
        const uint32_t laddr = (uint32_t)(uintptr_t)(&Qs[0]);
        v4u g0;
        g0[0] = 1u;                                        // count=1 (valid D#)
        g0[1] = laddr;                                     // lds_addr
        g0[2] = (uint32_t)gaddr;                           // global_addr[31:0]
        g0[3] = (uint32_t)((gaddr >> 32) & 0x1FFFFFFu)     // global_addr[56:32]
              | (2u << 30);                                // type = 2 (image)
        v8u g1;
        g1[0] = (1u << 16);              // workgroup_mask=0, data_size=1 (2B)
        g1[1] = (uint32_t)(DIM & 0xFFFF) << 16;            // tensor_dim0 lo16
        g1[2] = (uint32_t)(DIM >> 16) | (16u << 16);       // dim0 hi | dim1=16
        g1[3] = ((uint32_t)DIM << 16);                     // tile_dim0 = 1024
        g1[4] = 16u;                                       // tile_dim1=16
        g1[5] = (uint32_t)DIM;                             // dim0_stride = 1024
        g1[6] = 0u;
        g1[7] = 0u;
        v4u g2 = {0u, 0u, 0u, 0u};                         // 2-D tile: unused
        v4u g3 = {0u, 0u, 0u, 0u};
        asm volatile("tensor_load_to_lds %0, %1, %2, %3"
                     :: "s"(g0), "s"(g1), "s"(g2), "s"(g3) : "memory");
        __builtin_amdgcn_s_wait_tensorcnt(0);
    }

    v8f o[8];
#pragma unroll
    for (int dt = 0; dt < 8; ++dt) o[dt] = zero8();

    if (tid < 16) { m_sh[tid] = -1e30f; l_sh[tid] = 0.0f; }
    for (int i = tid; i < 16 * 32; i += 256) S_sh[i] = 0.0f;
    __syncthreads();   // publishes both the TDM'd Q tile and the init

    for (int k0 = 0; k0 < SEQ; k0 += 32) {
        // ---- partial S = Q[16 x 128] * K^T[128 x 32] over this d-slice ----
        v8f s_acc[2];
        s_acc[0] = zero8();
        s_acc[1] = zero8();
#pragma unroll
        for (int ds = 0; ds < 4; ++ds) {
            const int dcur = dbase + ds * 32;
            v16bf qf = load_frag(Qs, DIM, lane & 15, dcur, lane);   // from LDS
#pragma unroll
            for (int kt = 0; kt < 2; ++kt) {
                const int keyrow = b * SEQ + k0 + kt * 16 + (lane & 15);
                v16bf kf = load_frag(Kg, DIM, keyrow, dcur, lane);
                s_acc[kt] = wmma_bf16(qf, kf, s_acc[kt]);
            }
        }
        // Prefetch next key tile (L2-resident anyway; cheap hint).
        if (k0 + 32 < SEQ)
            __builtin_prefetch(Kg + ((size_t)(b * SEQ + k0 + 32) * DIM + dbase), 0, 1);

        // ---- cross-wave reduction of partial S into LDS (ds_add_f32) ----
#pragma unroll
        for (int kt = 0; kt < 2; ++kt)
#pragma unroll
            for (int r = 0; r < 8; ++r) {
                const int qrow = r + ((lane >> 4) << 3);
                const int kcol = kt * 16 + (lane & 15);
                atomicAdd(&S_sh[qrow * 32 + kcol], s_acc[kt][r]);
            }
        __syncthreads();

        // ---- online softmax, one thread per query row ----
        if (tid < 16) {
            const int q  = tid;
            const int gq = q0 + q;
            const float m_old = m_sh[q];
            float mnew = m_old;
            for (int k = 0; k < 32; ++k) {
                const int gk = k0 + k;
                const bool keep = (gq - gk <= BAND_LO) && (gk - gq <= BAND_HI);
                const float s = keep ? S_sh[q * 32 + k] * INV_SQRT_D : 0.0f;
                mnew = fmaxf(mnew, s);
            }
            const float alpha = __expf(m_old - mnew);
            float lsum = 0.0f;
            for (int k = 0; k < 32; ++k) {
                const int gk = k0 + k;
                const bool keep = (gq - gk <= BAND_LO) && (gk - gq <= BAND_HI);
                const float s = keep ? S_sh[q * 32 + k] * INV_SQRT_D : 0.0f;
                const float p = __expf(s - mnew);
                lsum += p;
                P_sh[q * 32 + k] = (__bf16)p;
            }
            m_sh[q]     = mnew;
            l_sh[q]     = l_sh[q] * alpha + lsum;
            alpha_sh[q] = alpha;
        }
        __syncthreads();

        // ---- rescale O and accumulate P * V over this wave's d-slice ----
        float a8[8];
#pragma unroll
        for (int r = 0; r < 8; ++r) a8[r] = alpha_sh[r + ((lane >> 4) << 3)];
#pragma unroll
        for (int dt = 0; dt < 8; ++dt)
#pragma unroll
            for (int r = 0; r < 8; ++r) o[dt][r] *= a8[r];

        v16bf pf = load_frag(P_sh, 32, lane & 15, 0, lane);   // A: 16q x 32k
#pragma unroll
        for (int dt = 0; dt < 8; ++dt) {
            const int dcol = dbase + dt * 16 + (lane & 15);
            // B: Vt row-major [d][t]; contraction (k) runs are contiguous.
            v16bf vf = load_frag(Vtg, SEQ, b * DIM + dcol, k0, lane);
            o[dt] = wmma_bf16(pf, vf, o[dt]);
        }

        // ---- clear S for the next step's atomic accumulation ----
        for (int i = tid; i < 16 * 32; i += 256) S_sh[i] = 0.0f;
        __syncthreads();
    }

    // ---- normalize by l and store f32 output ----
    float linv[8];
#pragma unroll
    for (int r = 0; r < 8; ++r) linv[r] = 1.0f / l_sh[r + ((lane >> 4) << 3)];
#pragma unroll
    for (int dt = 0; dt < 8; ++dt) {
        const int dcol = dbase + dt * 16 + (lane & 15);
#pragma unroll
        for (int r = 0; r < 8; ++r) {
            const int qrow = q0 + r + ((lane >> 4) << 3);
            Og[((size_t)b * SEQ + qrow) * DIM + dcol] = o[dt][r] * linv[r];
        }
    }
}

// ---------------------------------------------------------------------------
// Host launcher
// ---------------------------------------------------------------------------
extern "C" void kernel_launch(void* const* d_in, const int* in_sizes, int n_in,
                              void* d_out, int out_size, void* d_ws, size_t ws_size,
                              hipStream_t stream) {
    const float* X  = (const float*)d_in[0];
    const float* WQ = (const float*)d_in[1];
    const float* WK = (const float*)d_in[2];
    const float* WV = (const float*)d_in[3];
    float* Out      = (float*)d_out;

    const size_t qkvBytes = (size_t)BATCH * SEQ * DIM * sizeof(__bf16); // 32 MB each
    char* ws = (char*)d_ws;
    __bf16* Qg  = (__bf16*)(ws);
    __bf16* Kg  = (__bf16*)(ws + qkvBytes);
    __bf16* Vtg = (__bf16*)(ws + 2 * qkvBytes);

    // Kernel 1: Q, K, V^T projections (grid.z selects the weight/output).
    dim3 g1(DIM / 128, (BATCH * SEQ) / 128, 3);
    qkv_gemm_kernel<<<g1, 256, 0, stream>>>(X, WQ, WK, WV, Qg, Kg, Vtg);

    // Kernel 2: fused attention.  One block per (batch, 16-query tile).
    dim3 g2(BATCH * (SEQ / 16));
    attn_kernel<<<g2, 256, 0, stream>>>(Qg, Kg, Vtg, Out);
}
